// TimeAwareLSTMModel_21303037788602
// MI455X (gfx1250) — matrix operand order repaired and checked by reading, not compiled, not run on hardware
//
#include <hip/hip_runtime.h>

// ---------------------------------------------------------------------------
// Time-aware LSTM for MI455X (gfx1250, wave32, WMMA bf16 16x16x32)
// ---------------------------------------------------------------------------
#define VOCAB  20000
#define EMBED  256
#define HIDDEN 512
#define BB     256
#define TT     512
#define NCOLS  2560   // 4*HIDDEN (i,f,o,g) + HIDDEN (decay)
#define NCHUNK 24     // 8 x-chunks (K=256) + 16 h-chunks (K=512), 32-K each

typedef __attribute__((ext_vector_type(16))) __bf16 v16bf;
typedef __attribute__((ext_vector_type(8)))  float  v8f;
typedef __attribute__((ext_vector_type(4)))  float  f32x4;   // trivially-copyable 16B chunk

union Frag {
    v16bf v;
    f32x4 f4[2];
};

struct FragSet {          // one K-chunk worth of operands: A tile + five B tiles
    Frag A, B0, B1, B2, B3, B4;
};

__device__ __forceinline__ float sigmoidf_(float x) { return 1.0f / (1.0f + __expf(-x)); }
__device__ __forceinline__ float softplusf_(float x) {
    return (x > 20.0f) ? x : log1pf(__expf(x));
}

// ---------------------------------------------------------------------------
// One-time preprocessing kernels
// ---------------------------------------------------------------------------

// Gather embedding rows and convert to bf16: Xbf[b][t][e] = bf16(emb[X[b,t]][e])
__global__ void k_embed(const int* __restrict__ X, const float* __restrict__ emb,
                        __bf16* __restrict__ Xbf) {
    size_t i   = (size_t)blockIdx.x * blockDim.x + threadIdx.x;   // over B*T*EMBED
    size_t tok = i >> 8;           // EMBED == 256
    int    e   = (int)(i & 255);
    int    idx = X[tok];
    Xbf[i] = (__bf16)emb[(size_t)idx * EMBED + e];
}

// Pack [W_w | Wd_w] column-major (K contiguous) in bf16: Wpack[n][k], K = EMBED
__global__ void k_pack_w(const float* __restrict__ W_w, const float* __restrict__ Wd_w,
                         __bf16* __restrict__ Wpack) {
    int i = blockIdx.x * blockDim.x + threadIdx.x;   // over NCOLS*EMBED
    int n = i >> 8;
    int k = i & 255;
    float v = (n < 4 * HIDDEN) ? W_w[(size_t)k * (4 * HIDDEN) + n]
                               : Wd_w[(size_t)k * HIDDEN + (n - 4 * HIDDEN)];
    Wpack[(size_t)n * EMBED + k] = (__bf16)v;
}

// Pack [U_w | Ud_w] column-major (K contiguous) in bf16: Upack[n][k], K = HIDDEN
__global__ void k_pack_u(const float* __restrict__ U_w, const float* __restrict__ Ud_w,
                         __bf16* __restrict__ Upack) {
    int i = blockIdx.x * blockDim.x + threadIdx.x;   // over NCOLS*HIDDEN
    int n = i >> 9;
    int k = i & 511;
    float v = (n < 4 * HIDDEN) ? U_w[(size_t)k * (4 * HIDDEN) + n]
                               : Ud_w[(size_t)k * HIDDEN + (n - 4 * HIDDEN)];
    Upack[(size_t)n * HIDDEN + k] = (__bf16)v;
}

__global__ void k_init(float* __restrict__ c, __bf16* __restrict__ hbf) {
    int i = blockIdx.x * blockDim.x + threadIdx.x;   // over BB*HIDDEN
    c[i]   = 0.0f;
    hbf[i] = (__bf16)0.0f;
}

// ---------------------------------------------------------------------------
// Manually scheduled fragment loads: volatile asm keeps issue order, and the
// explicit s_wait_loadcnt (with tied operands) is the only wait in the chain.
// ---------------------------------------------------------------------------
__device__ __forceinline__ void gld(f32x4& d, const f32x4* p) {
    asm volatile("global_load_b128 %0, %1, off" : "=v"(d) : "v"(p));
}

// Issue all 12 b128 loads for chunk i (A:2, B:10), no wait.
// Chunks 0..7  : x-projection, A from Xbf row, B from Wpack (Kdim = EMBED)
// Chunks 8..23 : recurrence,   A from h row,   B from Upack (Kdim = HIDDEN)
__device__ __forceinline__ void issue_chunk(FragSet& s, int i,
                                            const __bf16* xrow, const __bf16* hrow,
                                            const __bf16* Wpack, const __bf16* Upack,
                                            int ncl, int half) {
    const f32x4* a0;
    const f32x4* b[5];
    if (i < 8) {
        const int kc   = i;
        const int koff = kc * 32 + half * 16;
        a0 = (const f32x4*)xrow + kc * 4;   // K = kc*32 + half*16 (.. +15)
#pragma unroll
        for (int g = 0; g < 5; ++g)
            b[g] = (const f32x4*)(Wpack + ((size_t)(g * HIDDEN + ncl)) * EMBED + koff);
    } else {
        const int kc   = i - 8;
        const int koff = kc * 32 + half * 16;
        a0 = (const f32x4*)hrow + kc * 4;
#pragma unroll
        for (int g = 0; g < 5; ++g)
            b[g] = (const f32x4*)(Upack + ((size_t)(g * HIDDEN + ncl)) * HIDDEN + koff);
    }
    gld(s.A.f4[0],  a0);        gld(s.A.f4[1],  a0 + 1);
    gld(s.B0.f4[0], b[0]);      gld(s.B0.f4[1], b[0] + 1);
    gld(s.B1.f4[0], b[1]);      gld(s.B1.f4[1], b[1] + 1);
    gld(s.B2.f4[0], b[2]);      gld(s.B2.f4[1], b[2] + 1);
    gld(s.B3.f4[0], b[3]);      gld(s.B3.f4[1], b[3] + 1);
    gld(s.B4.f4[0], b[4]);      gld(s.B4.f4[1], b[4] + 1);
}

// Wait until at most `next chunk` (12) loads are outstanding; ties all of the
// chunk's registers so consumers cannot be scheduled above the wait.
#define WAIT_FENCE(s, IMM)                                                        \
    asm volatile("s_wait_loadcnt " IMM                                            \
                 : "+v"((s).A.f4[0]),  "+v"((s).A.f4[1]),                         \
                   "+v"((s).B0.f4[0]), "+v"((s).B0.f4[1]),                        \
                   "+v"((s).B1.f4[0]), "+v"((s).B1.f4[1]),                        \
                   "+v"((s).B2.f4[0]), "+v"((s).B2.f4[1]),                        \
                   "+v"((s).B3.f4[0]), "+v"((s).B3.f4[1]),                        \
                   "+v"((s).B4.f4[0]), "+v"((s).B4.f4[1]))

__device__ __forceinline__ void consume_chunk(const FragSet& s,
                                              v8f& acc0, v8f& acc1, v8f& acc2,
                                              v8f& acc3, v8f& accd) {
    acc0 = __builtin_amdgcn_wmma_f32_16x16x32_bf16(false, s.A.v, false, s.B0.v,
                                                   (short)0, acc0, false, false);
    acc1 = __builtin_amdgcn_wmma_f32_16x16x32_bf16(false, s.A.v, false, s.B1.v,
                                                   (short)0, acc1, false, false);
    acc2 = __builtin_amdgcn_wmma_f32_16x16x32_bf16(false, s.A.v, false, s.B2.v,
                                                   (short)0, acc2, false, false);
    acc3 = __builtin_amdgcn_wmma_f32_16x16x32_bf16(false, s.A.v, false, s.B3.v,
                                                   (short)0, acc3, false, false);
    accd = __builtin_amdgcn_wmma_f32_16x16x32_bf16(false, s.A.v, false, s.B4.v,
                                                   (short)0, accd, false, false);
}

// ---------------------------------------------------------------------------
// One recurrent step. Grid = 64 blocks x 256 threads = 512 waves.
// Wave map: j = wid>>4 (hidden tile, fixed per 8-wave block -> B tiles shared
// through WGP$), m = wid&15 (batch tile, varies within block).
// Hand-pipelined: chunk k+1's 12 loads are in flight while chunk k's five
// WMMAs execute (s_wait_loadcnt 0xc steady state, 0x0 only at the tail).
// ---------------------------------------------------------------------------
__launch_bounds__(256)
__global__ void k_step(const __bf16* __restrict__ Xbf, int t,
                       const __bf16* __restrict__ hbf_in,
                       __bf16* __restrict__ hbf_out,
                       float* __restrict__ c,
                       float* __restrict__ hout,
                       const __bf16* __restrict__ Wpack,
                       const __bf16* __restrict__ Upack,
                       const float* __restrict__ W_b,  const float* __restrict__ U_b,
                       const float* __restrict__ Wd_b, const float* __restrict__ Ud_b,
                       const float* __restrict__ deltas) {
    const int lane = threadIdx.x & 31;
    const int wid  = (blockIdx.x * blockDim.x + threadIdx.x) >> 5;  // 0..511
    const int j    = wid >> 4;   // hidden tile 0..31 (constant within a block)
    const int m    = wid & 15;   // batch tile  0..15
    const int cl   = lane & 15;  // N column (B/C/D) or M row (A)
    const int half = lane >> 4;  // K-half selector

    v8f acc0 = {}, acc1 = {}, acc2 = {}, acc3 = {}, accd = {};

    const int ncl = j * 16 + cl;

    const __bf16* xrow = Xbf + (((size_t)(m * 16 + cl)) * TT + t) * EMBED + half * 16;
    const __bf16* hrow = hbf_in + (size_t)(m * 16 + cl) * HIDDEN + half * 16;

    // ---- hand-pipelined GEMM accumulation over 24 K-chunks of 32
    FragSet s0, s1;
    issue_chunk(s0, 0, xrow, hrow, Wpack, Upack, ncl, half);
#pragma unroll
    for (int i = 0; i < NCHUNK - 2; i += 2) {
        issue_chunk(s1, i + 1, xrow, hrow, Wpack, Upack, ncl, half);
        WAIT_FENCE(s0, "0xc");
        consume_chunk(s0, acc0, acc1, acc2, acc3, accd);
        issue_chunk(s0, i + 2, xrow, hrow, Wpack, Upack, ncl, half);
        WAIT_FENCE(s1, "0xc");
        consume_chunk(s1, acc0, acc1, acc2, acc3, accd);
    }
    // tail: chunks NCHUNK-2 (already in s0) and NCHUNK-1
    issue_chunk(s1, NCHUNK - 1, xrow, hrow, Wpack, Upack, ncl, half);
    WAIT_FENCE(s0, "0xc");
    consume_chunk(s0, acc0, acc1, acc2, acc3, accd);
    WAIT_FENCE(s1, "0x0");   // full drain: keeps compiler's epilogue waits sound
    consume_chunk(s1, acc0, acc1, acc2, acc3, accd);

    // ---- epilogue: gates + time-decayed cell update
    const float bi = W_b[ncl]              + U_b[ncl];
    const float bf = W_b[HIDDEN + ncl]     + U_b[HIDDEN + ncl];
    const float bo = W_b[2 * HIDDEN + ncl] + U_b[2 * HIDDEN + ncl];
    const float bg = W_b[3 * HIDDEN + ncl] + U_b[3 * HIDDEN + ncl];
    const float bd = Wd_b[ncl]             + Ud_b[ncl];

#pragma unroll
    for (int v = 0; v < 8; ++v) {
        const int    b   = m * 16 + half * 8 + v;     // C/D layout: lane half selects M+8
        const size_t off = (size_t)b * HIDDEN + ncl;
        const float  dt  = deltas[(size_t)b * TT + t];

        const float dval  = softplusf_(accd[v] + bd);
        const float gamma = __expf(-dval * dt);
        const float iv    = sigmoidf_(acc0[v] + bi);
        const float fv    = sigmoidf_(acc1[v] + bf);
        const float ov    = sigmoidf_(acc2[v] + bo);
        const float gv    = tanhf(acc3[v] + bg);

        const float cn = fv * gamma * c[off] + iv * gv;
        const float hn = ov * tanhf(cn);

        c[off]        = cn;
        hout[off]     = hn;
        hbf_out[off]  = (__bf16)hn;
    }
}

// ---------------------------------------------------------------------------
// logits[b] = h[b,:] . fc_w + fc_b
// ---------------------------------------------------------------------------
__global__ void k_logits(const float* __restrict__ h, const float* __restrict__ fc_w,
                         const float* __restrict__ fc_b, float* __restrict__ out) {
    __shared__ float red[256];
    const int b   = blockIdx.x;
    const int tid = threadIdx.x;   // 256 threads
    float s = h[(size_t)b * HIDDEN + tid] * fc_w[tid] +
              h[(size_t)b * HIDDEN + 256 + tid] * fc_w[256 + tid];
    red[tid] = s;
    __syncthreads();
    for (int o = 128; o > 0; o >>= 1) {
        if (tid < o) red[tid] += red[tid + o];
        __syncthreads();
    }
    if (tid == 0) out[b] = red[0] + fc_b[0];
}

// ---------------------------------------------------------------------------
extern "C" void kernel_launch(void* const* d_in, const int* in_sizes, int n_in,
                              void* d_out, int out_size, void* d_ws, size_t ws_size,
                              hipStream_t stream) {
    const int*   X      = (const int*)d_in[0];
    // d_in[1] = M (all-ones mask, unused by the reference computation)
    const float* deltas = (const float*)d_in[2];
    const float* emb    = (const float*)d_in[3];
    const float* W_w    = (const float*)d_in[4];
    const float* W_b    = (const float*)d_in[5];
    const float* U_w    = (const float*)d_in[6];
    const float* U_b    = (const float*)d_in[7];
    const float* Wd_w   = (const float*)d_in[8];
    const float* Wd_b   = (const float*)d_in[9];
    const float* Ud_w   = (const float*)d_in[10];
    const float* Ud_b   = (const float*)d_in[11];
    const float* fc_w   = (const float*)d_in[12];
    const float* fc_b   = (const float*)d_in[13];

    // Workspace layout (all chunks 256B-aligned; total ~72 MB)
    char*   ws    = (char*)d_ws;
    __bf16* Xbf   = (__bf16*)ws;  ws += (size_t)BB * TT * EMBED * 2;   // 64 MiB
    __bf16* Wpack = (__bf16*)ws;  ws += (size_t)NCOLS * EMBED * 2;     // 1.25 MiB
    __bf16* Upack = (__bf16*)ws;  ws += (size_t)NCOLS * HIDDEN * 2;    // 2.5 MiB
    __bf16* hbf0  = (__bf16*)ws;  ws += (size_t)BB * HIDDEN * 2;       // 256 KiB
    __bf16* hbf1  = (__bf16*)ws;  ws += (size_t)BB * HIDDEN * 2;       // 256 KiB
    float*  c     = (float*)ws;   ws += (size_t)BB * HIDDEN * 4;       // 512 KiB

    float* out  = (float*)d_out;
    float* hout = out + BB;   // output h region follows logits (return order)

    k_embed <<<(size_t)BB * TT * EMBED / 256, 256, 0, stream>>>(X, emb, Xbf);
    k_pack_w<<<NCOLS * EMBED / 256, 256, 0, stream>>>(W_w, Wd_w, Wpack);
    k_pack_u<<<NCOLS * HIDDEN / 256, 256, 0, stream>>>(U_w, Ud_w, Upack);
    k_init  <<<BB * HIDDEN / 256, 256, 0, stream>>>(c, hbf0);

    for (int t = 0; t < TT; ++t) {
        const __bf16* hin = (t & 1) ? hbf1 : hbf0;
        __bf16*       ho  = (t & 1) ? hbf0 : hbf1;
        k_step<<<64, 256, 0, stream>>>(Xbf, t, hin, ho, c, hout, Wpack, Upack,
                                       W_b, U_b, Wd_b, Ud_b, deltas);
    }

    k_logits<<<BB, 256, 0, stream>>>(hout, fc_w, fc_b, out);
}
